// GlobalAttention_1_48515950576285
// MI455X (gfx1250) — compile-verified, matrix operand order checked by reading, not run on hardware
//
#include <hip/hip_runtime.h>

// ---------------------------------------------------------------------------
// GlobalAttention for MI455X (gfx1250): energy = P P^T (symmetric, upper
// triangle only + mirrored stores), softmax(rowmax - e), out = att * P,
// y = gamma*out + x.   P = x viewed as (C=1024, N=36864) f32.
// GEMMs on v_wmma_f32_16x16x32_bf16 (wave32), 32x64 wave tiles (8 WMMAs/step),
// double-buffered LDS + register prefetch, v_cvt_pk_bf16_f32 conversions.
// ---------------------------------------------------------------------------

#define C_DIM 1024
#define N_DIM 36864            // 192*192
#define KSPLIT 8
#define KCHUNK (N_DIM / KSPLIT)  // 4608, multiple of 32
#define CTILES (C_DIM / 128)     // 8
#define NTRI   (CTILES * (CTILES + 1) / 2)  // 36 upper-triangle tiles

typedef __bf16 bf16_t;
typedef __attribute__((ext_vector_type(2)))  bf16_t bf16x2_t;
typedef __attribute__((ext_vector_type(16))) bf16_t v16bf;
typedef __attribute__((ext_vector_type(8)))  float  v8f;

union FragBF {
    v16bf    v;
    unsigned u[8];
};

// 2x f32 -> packed bf16x2 in one VALU op (v_cvt_pk_bf16_f32)
__device__ __forceinline__ unsigned pack_bf16x2(float lo, float hi) {
#if __has_builtin(__builtin_amdgcn_cvt_pk_bf16_f32)
    bf16x2_t p = __builtin_amdgcn_cvt_pk_bf16_f32(lo, hi);
    return __builtin_bit_cast(unsigned, p);
#else
    bf16x2_t p;
    p.x = (bf16_t)lo;
    p.y = (bf16_t)hi;
    return __builtin_bit_cast(unsigned, p);
#endif
}

// K-index -> packed-uint index in a 32-deep bf16 fragment row.
// 16-bit A/B layout (ISA 7.12.2): VGPR j holds K = 16*(j>>2) + 8*laneHalf + 2*(j&3)
__device__ __forceinline__ int frag_kpair(int j, int kh) {
    return ((j >> 2) << 3) + (kh << 2) + (j & 3);
}

// ---------------------------------------------------------------------------
// Kernel 1: partial energy[c][d] = sum_{k in chunk} P[c][k]*P[d][k]
// 128x128 block tile; 8 waves 4x2; 32x64 per wave (8 WMMAs/K-step).
// Upper-triangle tiles only; off-diagonal tiles mirror-store the transpose.
// Grid (NTRI, 1, KSPLIT).
// ---------------------------------------------------------------------------
__global__ __launch_bounds__(256) void ga_energy_kernel(const float* __restrict__ x,
                                                        float* __restrict__ energy_part) {
    __shared__ unsigned ldsA[2][128 * 17];  // c-rows x 32 K, bf16 pairs (+pad)
    __shared__ unsigned ldsB[2][128 * 17];  // d-rows x 32 K

    // decode upper-triangle tile (ti <= tj)
    int l = blockIdx.x, ti = 0, rem = CTILES;
    while (l >= rem) { l -= rem; --rem; ++ti; }
    const int tj = ti + l;
    const int c0 = ti * 128;
    const int d0 = tj * 128;

    const int t    = threadIdx.x;
    const int lane = t & 31;
    const int wave = t >> 5;
    const int m_off = (wave >> 1) * 32;     // 0,32,64,96
    const int n_off = (wave & 1) * 64;      // 0,64

    const int kbeg = blockIdx.z * KCHUNK;
    const int kend = kbeg + KCHUNK;

    // staging: A 128x32 and B 128x32 -> 16 f32/thread each
    const int sr = t >> 1, sh = (t & 1) * 16;
    const float* aptr = x + (size_t)(c0 + sr) * N_DIM + sh;
    const float* bptr = x + (size_t)(d0 + sr) * N_DIM + sh;

    const int m  = lane & 15;
    const int kh = lane >> 4;

    v8f acc[2][4];
#pragma unroll
    for (int im = 0; im < 2; ++im)
#pragma unroll
        for (int in = 0; in < 4; ++in) acc[im][in] = (v8f){};

    float4 ra[4], rb[4];
#pragma unroll
    for (int j = 0; j < 4; ++j) {
        ra[j] = *(const float4*)(aptr + kbeg + 4 * j);
        rb[j] = *(const float4*)(bptr + kbeg + 4 * j);
    }

    int buf = 0;
    for (int kt = kbeg; kt < kend; kt += 32, buf ^= 1) {
        // ---- commit staged registers to LDS[buf] ----
        unsigned* pa = &ldsA[buf][sr * 17 + (sh >> 1)];
        unsigned* pb = &ldsB[buf][sr * 17 + (sh >> 1)];
#pragma unroll
        for (int j = 0; j < 4; ++j) {
            pa[2 * j]     = pack_bf16x2(ra[j].x, ra[j].y);
            pa[2 * j + 1] = pack_bf16x2(ra[j].z, ra[j].w);
            pb[2 * j]     = pack_bf16x2(rb[j].x, rb[j].y);
            pb[2 * j + 1] = pack_bf16x2(rb[j].z, rb[j].w);
        }
        __syncthreads();

        // ---- overlap: fetch next tile into registers ----
        const int kn = kt + 32;
        if (kn < kend) {
#pragma unroll
            for (int j = 0; j < 4; ++j) {
                ra[j] = *(const float4*)(aptr + kn + 4 * j);
                rb[j] = *(const float4*)(bptr + kn + 4 * j);
            }
            __builtin_prefetch((const char*)(aptr + kn) + 256, 0, 3);
            __builtin_prefetch((const char*)(bptr + kn) + 256, 0, 3);
        }

        // ---- compute from LDS[buf]: 2 A-frags x 4 B-frags = 8 WMMAs ----
        FragBF fa[2], fb[4];
#pragma unroll
        for (int j = 0; j < 8; ++j) {
            const int kp = frag_kpair(j, kh);
            fa[0].u[j] = ldsA[buf][(m_off + m) * 17 + kp];
            fa[1].u[j] = ldsA[buf][(m_off + 16 + m) * 17 + kp];
            fb[0].u[j] = ldsB[buf][(n_off + m) * 17 + kp];
            fb[1].u[j] = ldsB[buf][(n_off + 16 + m) * 17 + kp];
            fb[2].u[j] = ldsB[buf][(n_off + 32 + m) * 17 + kp];
            fb[3].u[j] = ldsB[buf][(n_off + 48 + m) * 17 + kp];
        }
#pragma unroll
        for (int im = 0; im < 2; ++im)
#pragma unroll
            for (int in = 0; in < 4; ++in)
                acc[im][in] = __builtin_amdgcn_wmma_f32_16x16x32_bf16(
                    false, fa[im].v, false, fb[in].v, (short)0, acc[im][in], false, false);
        __syncthreads();
    }

    // ---- store C/D layout -> partial energy plane (+ mirrored tile) ----
    float* ep = energy_part + (size_t)blockIdx.z * C_DIM * C_DIM;
    const int n = lane & 15;
#pragma unroll
    for (int im = 0; im < 2; ++im)
#pragma unroll
        for (int in = 0; in < 4; ++in)
#pragma unroll
            for (int v = 0; v < 8; ++v) {
                const int row = c0 + m_off + im * 16 + v + 8 * kh;
                const int col = d0 + n_off + in * 16 + n;
                const float val = acc[im][in][v];
                ep[(size_t)row * C_DIM + col] = val;
                if (ti != tj) ep[(size_t)col * C_DIM + row] = val;   // symmetry
            }
}

// ---------------------------------------------------------------------------
// Kernel 2: attention[c][d] = exp(rowmin_c - e) / sum  (== softmax(rowmax-e))
// Sums the KSPLIT partial energy planes. One block per row; bf16 output.
// ---------------------------------------------------------------------------
__global__ __launch_bounds__(256) void ga_softmax_kernel(const float* __restrict__ energy_part,
                                                         unsigned short* __restrict__ att) {
    __shared__ float red[256];
    const int row  = blockIdx.x;
    const int t    = threadIdx.x;
    const int base = t * 4;

    float e[4] = {0.f, 0.f, 0.f, 0.f};
#pragma unroll
    for (int z = 0; z < KSPLIT; ++z) {
        const float4 ev = *(const float4*)(energy_part + (size_t)z * C_DIM * C_DIM +
                                           (size_t)row * C_DIM + base);
        e[0] += ev.x; e[1] += ev.y; e[2] += ev.z; e[3] += ev.w;
    }
    float lmin = fminf(fminf(e[0], e[1]), fminf(e[2], e[3]));

    red[t] = lmin;
    __syncthreads();
    for (int s = 128; s > 0; s >>= 1) {
        if (t < s) red[t] = fminf(red[t], red[t + s]);
        __syncthreads();
    }
    const float rmin = red[0];
    __syncthreads();

    float p[4];
    float lsum = 0.f;
#pragma unroll
    for (int i = 0; i < 4; ++i) {
        p[i] = __expf(rmin - e[i]);
        lsum += p[i];
    }
    red[t] = lsum;
    __syncthreads();
    for (int s = 128; s > 0; s >>= 1) {
        if (t < s) red[t] += red[t + s];
        __syncthreads();
    }
    const float inv = 1.0f / red[0];

    unsigned* out = (unsigned*)(att + (size_t)row * C_DIM + base);
    out[0] = pack_bf16x2(p[0] * inv, p[1] * inv);
    out[1] = pack_bf16x2(p[2] * inv, p[3] * inv);
}

// ---------------------------------------------------------------------------
// Kernel 3: y[c][n] = gamma * (sum_d att[c][d] * P[d][n]) + x[c][n]
// 128(c) x 128(n) block tile; 8 waves 4x2; 32x64 per wave; K = 1024.
// B staged transposed: thread owns one n-column, 16 consecutive k.
// Grid (288, 8).
// ---------------------------------------------------------------------------
__global__ __launch_bounds__(256) void ga_out_kernel(const float* __restrict__ x,
                                                     const unsigned short* __restrict__ att,
                                                     const float* __restrict__ gamma,
                                                     float* __restrict__ y) {
    __shared__ unsigned       ldsA[2][128 * 17];  // att: c-rows x 32 d (bf16 pairs)
    __shared__ unsigned short ldsBT[2][128 * 34]; // P^T: n-rows x 32 d (+pad, uint-aligned)

    const int t    = threadIdx.x;
    const int lane = t & 31;
    const int wave = t >> 5;
    const int m_off = (wave >> 1) * 32;
    const int n_off = (wave & 1) * 64;

    const int n0 = blockIdx.x * 128;
    const int c0 = blockIdx.y * 128;

    // A staging: 128 rows x 32 d bf16 -> 8 uints/thread
    const int ar = t >> 1, ah = (t & 1) * 16;
    const unsigned* aptr0 = (const unsigned*)(att + (size_t)(c0 + ar) * C_DIM + ah);
    // B staging (transposed): thread owns n-column bn, k in [bkq, bkq+16)
    const int bn = t & 127, bkq = (t >> 7) * 16;
    const float* bcol = x + (size_t)bkq * N_DIM + n0 + bn;   // + (kt + j)*N_DIM

    const int m  = lane & 15;
    const int kh = lane >> 4;

    v8f acc[2][4];
#pragma unroll
    for (int im = 0; im < 2; ++im)
#pragma unroll
        for (int in = 0; in < 4; ++in) acc[im][in] = (v8f){};

    uint4 raw[2];
    float rbv[16];
    raw[0] = *(const uint4*)(aptr0);
    raw[1] = *(const uint4*)(aptr0 + 4);
#pragma unroll
    for (int j = 0; j < 16; ++j) rbv[j] = bcol[(size_t)j * N_DIM];

    int buf = 0;
    for (int kt = 0; kt < C_DIM; kt += 32, buf ^= 1) {
        // ---- commit staged registers to LDS[buf] ----
        unsigned* pa = &ldsA[buf][ar * 17 + (ah >> 1)];
        pa[0] = raw[0].x; pa[1] = raw[0].y; pa[2] = raw[0].z; pa[3] = raw[0].w;
        pa[4] = raw[1].x; pa[5] = raw[1].y; pa[6] = raw[1].z; pa[7] = raw[1].w;
        unsigned* pbt = (unsigned*)&ldsBT[buf][bn * 34 + bkq]; // bn*68 + bkq*2 bytes, 4-aligned
#pragma unroll
        for (int j = 0; j < 8; ++j)
            pbt[j] = pack_bf16x2(rbv[2 * j], rbv[2 * j + 1]);
        __syncthreads();

        // ---- overlap: fetch next K tile ----
        const int kn = kt + 32;
        if (kn < C_DIM) {
            raw[0] = *(const uint4*)(aptr0 + (kn >> 1));       // kn/2 uints = kn bf16
            raw[1] = *(const uint4*)(aptr0 + (kn >> 1) + 4);
#pragma unroll
            for (int j = 0; j < 16; ++j) rbv[j] = bcol[(size_t)(kn + j) * N_DIM];
            __builtin_prefetch((const char*)(bcol + (size_t)kn * N_DIM), 0, 3);
        }

        // ---- compute from LDS[buf]: 2 A-frags x 4 B-frags = 8 WMMAs ----
        FragBF fa[2], fb[4];
        const unsigned* rB0 = (const unsigned*)&ldsBT[buf][(n_off + m) * 34];
        const unsigned* rB1 = (const unsigned*)&ldsBT[buf][(n_off + 16 + m) * 34];
        const unsigned* rB2 = (const unsigned*)&ldsBT[buf][(n_off + 32 + m) * 34];
        const unsigned* rB3 = (const unsigned*)&ldsBT[buf][(n_off + 48 + m) * 34];
#pragma unroll
        for (int j = 0; j < 8; ++j) {
            const int kp = frag_kpair(j, kh);
            fa[0].u[j] = ldsA[buf][(m_off + m) * 17 + kp];
            fa[1].u[j] = ldsA[buf][(m_off + 16 + m) * 17 + kp];
            fb[0].u[j] = rB0[kp];
            fb[1].u[j] = rB1[kp];
            fb[2].u[j] = rB2[kp];
            fb[3].u[j] = rB3[kp];
        }
#pragma unroll
        for (int im = 0; im < 2; ++im)
#pragma unroll
            for (int in = 0; in < 4; ++in)
                acc[im][in] = __builtin_amdgcn_wmma_f32_16x16x32_bf16(
                    false, fa[im].v, false, fb[in].v, (short)0, acc[im][in], false, false);
        __syncthreads();
    }

    // ---- epilogue: y = gamma*out + x (flat index c*N + n) ----
    const float g = gamma[0];
    const int n = lane & 15;
#pragma unroll
    for (int im = 0; im < 2; ++im)
#pragma unroll
        for (int in = 0; in < 4; ++in)
#pragma unroll
            for (int v = 0; v < 8; ++v) {
                const size_t idx = (size_t)(c0 + m_off + im * 16 + v + 8 * kh) * N_DIM +
                                   n0 + n_off + in * 16 + n;
                y[idx] = g * acc[im][in][v] + x[idx];
            }
}

// ---------------------------------------------------------------------------
extern "C" void kernel_launch(void* const* d_in, const int* in_sizes, int n_in,
                              void* d_out, int out_size, void* d_ws, size_t ws_size,
                              hipStream_t stream) {
    const float* x     = (const float*)d_in[0];
    const float* gamma = (const float*)d_in[1];
    float*       y     = (float*)d_out;

    float*          energy_part = (float*)d_ws;                     // KSPLIT * 4 MB
    unsigned short* att = (unsigned short*)((char*)d_ws +
                          (size_t)KSPLIT * C_DIM * C_DIM * sizeof(float)); // +2 MB

    dim3 g1(NTRI, 1, KSPLIT);                   // 36 x 1 x 8 = 288 blocks
    ga_energy_kernel<<<g1, 256, 0, stream>>>(x, energy_part);

    ga_softmax_kernel<<<dim3(C_DIM), 256, 0, stream>>>(energy_part, att);

    dim3 g3(N_DIM / 128, C_DIM / 128);          // 288 x 8
    ga_out_kernel<<<g3, 256, 0, stream>>>(x, att, gamma, y);
}
